// MultiHeadSelfAttention_39573828665744
// MI455X (gfx1250) — matrix operand order repaired and checked by reading, not compiled, not run on hardware
//
#include <hip/hip_runtime.h>

// MI455X / gfx1250: wave32, WMMA f32_16x16x32_f16 (f32 accumulate), TDM async tiles.
// MHA forward: B=4, S=2048, D=1024, H=16, DK=64.
// Pipeline: cvt fp32->fp16 -> QKV gemm (WMMA + TDM double-buffer) -> flash attention
// (WMMA + TDM K-tiles) -> O gemm (WMMA + TDM) -> fp32 out.
// Workspace (f16): xh[8M] wq/wk/wv/wo[1M each] Q/K/V[8M each] A[8M] = ~92 MB.

typedef __attribute__((ext_vector_type(16))) _Float16 v16h;
typedef __attribute__((ext_vector_type(8)))  _Float16 v8h;
typedef __attribute__((ext_vector_type(8)))  float    v8f;
typedef __attribute__((ext_vector_type(4)))  float    v4f;

typedef __attribute__((ext_vector_type(4))) unsigned int tdm_v4u;
typedef __attribute__((ext_vector_type(8))) int          tdm_v8i;
typedef __attribute__((ext_vector_type(4))) int          tdm_v4i;

#define B_  4
#define S_  2048
#define D_  1024
#define H_  16
#define DK_ 64

#if defined(__has_builtin)
#  if __has_builtin(__builtin_amdgcn_tensor_load_to_lds) && \
      __has_builtin(__builtin_amdgcn_s_wait_tensorcnt)
#    define HAVE_TDM 1
#  else
#    define HAVE_TDM 0
#  endif
#else
#  define HAVE_TDM 0
#endif

static __device__ __forceinline__ v16h cat16(v8h lo, v8h hi) {
  return __builtin_shufflevector(lo, hi, 0,1,2,3,4,5,6,7,8,9,10,11,12,13,14,15);
}
static __device__ __forceinline__ v8f wmma16(v16h a, v16h b, v8f c) {
  return __builtin_amdgcn_wmma_f32_16x16x32_f16(false, a, false, b, (short)0, c, false, false);
}

// A fragment: 16(M) x 32(K) f16, row-major [m][k], stride ld. Lane: m=lane&15,
// hl=lane>>4; slots 0..7 = K hl*8.., slots 8..15 = K 16+hl*8.. -> 2x ds_load_b128.
static __device__ __forceinline__ v16h frag_a(const _Float16* p, int ld, int lane) {
  const int m = lane & 15, hl = lane >> 4;
  v8h lo = *(const v8h*)(p + m * ld + hl * 8);
  v8h hi = *(const v8h*)(p + m * ld + 16 + hl * 8);
  return cat16(lo, hi);
}
// B fragment: 32(K) x 16(N), source stored TRANSPOSED bt[n][k], stride ld.
// Lane n=lane&15, hl=lane>>4; slots 0..7 = K hl*8.., slots 8..15 = K 16+hl*8.. (matches frag_a).
static __device__ __forceinline__ v16h frag_b(const _Float16* bt, int ld, int lane) {
  const int n = lane & 15, hl = lane >> 4;
  v8h lo = *(const v8h*)(bt + n * ld + hl * 8);
  v8h hi = *(const v8h*)(bt + n * ld + 16 + hl * 8);
  return cat16(lo, hi);
}

#if HAVE_TDM
// One TDM 2D tile load: global (row-major, stride_d0 in f16 elements) -> LDS with
// per-row padding so rows land on our padded LDS strides (ISA 08 Tensor D# packing).
static __device__ __forceinline__ void tdm_load_2d(
    const void* gaddr, unsigned lds_off_bytes,
    unsigned tile_d0, unsigned tile_d1, unsigned long stride_d0,
    unsigned pad_interval_enc, unsigned pad_amount_enc) {
  const unsigned long ga = (unsigned long)gaddr;
  tdm_v4u g0;
  g0[0] = 1u;                                       // count=1, user desc, no gather
  g0[1] = lds_off_bytes;                            // lds_addr (bytes)
  g0[2] = (unsigned)(ga & 0xffffffffu);             // global_addr[31:0]
  g0[3] = (unsigned)((ga >> 32) & 0x1ffffffu) | 0x80000000u;  // addr[56:32] | type=2
  tdm_v8i g1;
  unsigned w0 = (1u << 16)                          // data_size = 1 -> 2 bytes (f16)
              | (1u << 20)                          // pad_enable
              | (pad_interval_enc << 22)
              | (pad_amount_enc << 25);
  g1[0] = (int)w0;
  g1[1] = (int)((tile_d0 & 0xffffu) << 16);         // tensor_dim0[15:0] @ bits 63:48
  g1[2] = (int)(((tile_d0 >> 16) & 0xffffu) | ((tile_d1 & 0xffffu) << 16));
  g1[3] = (int)(((tile_d1 >> 16) & 0xffffu) | ((tile_d0 & 0xffffu) << 16)); // tile_dim0
  g1[4] = (int)(tile_d1 & 0xffffu);                 // tile_dim1 (tile_dim2 = 0)
  g1[5] = (int)(unsigned)(stride_d0 & 0xffffffffu); // tensor_dim0_stride[31:0]
  g1[6] = (int)(unsigned)((stride_d0 >> 32) & 0xffffu);
  g1[7] = 0;
  tdm_v4i g2 = {0, 0, 0, 0};
  tdm_v4i g3 = {0, 0, 0, 0};
#if __clang_major__ >= 23
  tdm_v8i gx = {0, 0, 0, 0, 0, 0, 0, 0};
  __builtin_amdgcn_tensor_load_to_lds(g0, g1, g2, g3, gx, 0);
#else
  __builtin_amdgcn_tensor_load_to_lds(g0, g1, g2, g3, 0);
#endif
}
#endif

// ---------------------------------------------------------------- conversion
// Vectorized: 32B float loads -> 16B f16 stores, 8 elements per thread per trip.
__global__ void cvt_f32_f16(const float* __restrict__ src, _Float16* __restrict__ dst, int n8) {
  int i = blockIdx.x * blockDim.x + threadIdx.x;
  const int stride = gridDim.x * blockDim.x;
  for (; i < n8; i += stride) {
    const long o = (long)i * 8;
    v4f f0 = *(const v4f*)(src + o);
    v4f f1 = *(const v4f*)(src + o + 4);
    v8h hh;
    hh[0] = (_Float16)f0[0]; hh[1] = (_Float16)f0[1];
    hh[2] = (_Float16)f0[2]; hh[3] = (_Float16)f0[3];
    hh[4] = (_Float16)f1[0]; hh[5] = (_Float16)f1[1];
    hh[6] = (_Float16)f1[2]; hh[7] = (_Float16)f1[3];
    *(v8h*)(dst + o) = hh;
  }
}

// ---------------------------------------------------------------- GEMM core
static __device__ __forceinline__ void mma_step(const _Float16* la, const _Float16* lb,
                                                int wy, int wx, int lane, v8f c[2][2]) {
  v16h a0 = frag_a(la + (wy * 32) * 48, 48, lane);
  v16h a1 = frag_a(la + (wy * 32 + 16) * 48, 48, lane);
  v16h b0 = frag_b(lb + (wx * 32) * 48, 48, lane);
  v16h b1 = frag_b(lb + (wx * 32 + 16) * 48, 48, lane);
  c[0][0] = wmma16(a0, b0, c[0][0]);
  c[0][1] = wmma16(a0, b1, c[0][1]);
  c[1][0] = wmma16(a1, b0, c[1][0]);
  c[1][1] = wmma16(a1, b1, c[1][1]);
}

#if HAVE_TDM
static __device__ __forceinline__ void tdm_issue_pair(
    const _Float16* A, const _Float16* W, int m0, int n0, int kt,
    unsigned lds_a_off, unsigned lds_b_off) {
  const int k0 = kt * 32;
  // rows: 16 DW + 8 DW pad -> LDS stride 48 f16
  tdm_load_2d(A + (long)m0 * D_ + k0, lds_a_off, 32, 64, D_, 3, 7);
  tdm_load_2d(W + (long)n0 * D_ + k0, lds_b_off, 32, 128, D_, 3, 7);
}
#endif

// C[64x128] block tile; 8 waves (2x4), each wave 32x32 via 2x2 WMMA tiles.
// TDM double buffering, manually unrolled x2 so each half is straight-line code
// bound to a compile-time LDS buffer (no accumulator copies / pointer selects).
static __device__ __forceinline__ void gemm_core(
    const _Float16* __restrict__ A, const _Float16* __restrict__ W,
    int m0, int n0, _Float16* lds_a, _Float16* lds_bt, v8f c[2][2]) {
  const int tid = threadIdx.x, lane = tid & 31, wave = tid >> 5;
  const int wy = wave >> 2, wx = wave & 3;
  v8f z = {};
  c[0][0] = z; c[0][1] = z; c[1][0] = z; c[1][1] = z;
  const int KT = D_ / 32;   // 32, even

#if HAVE_TDM
  const unsigned aoff0 = (unsigned)(size_t)lds_a;       // low 32 bits = LDS byte offset
  const unsigned boff0 = (unsigned)(size_t)lds_bt;
  const unsigned aoff1 = aoff0 + 64 * 48 * 2;
  const unsigned boff1 = boff0 + 128 * 48 * 2;
  if (wave == 0) tdm_issue_pair(A, W, m0, n0, 0, aoff0, boff0);
  for (int kt = 0; kt < KT; kt += 2) {
    // ---- half 0: compute buf0 (tile kt), TDM fills buf1 (tile kt+1)
    if (wave == 0) {
      tdm_issue_pair(A, W, m0, n0, kt + 1, aoff1, boff1);
      __builtin_amdgcn_s_wait_tensorcnt(2);     // pair kt landed, kt+1 in flight
    }
    __syncthreads();
    mma_step(lds_a, lds_bt, wy, wx, lane, c);
    __syncthreads();
    // ---- half 1: compute buf1 (tile kt+1), TDM fills buf0 (tile kt+2)
    if (wave == 0) {
      if (kt + 2 < KT) {
        tdm_issue_pair(A, W, m0, n0, kt + 2, aoff0, boff0);
        __builtin_amdgcn_s_wait_tensorcnt(2);
      } else {
        __builtin_amdgcn_s_wait_tensorcnt(0);
      }
    }
    __syncthreads();
    mma_step(lds_a + 64 * 48, lds_bt + 128 * 48, wy, wx, lane, c);
    __syncthreads();
  }
#else
  for (int kt = 0; kt < KT; ++kt) {
    const int k0 = kt * 32;
    __syncthreads();
    {
      const int row = tid >> 2, ch = tid & 3;
      *(v8h*)&lds_a[row * 48 + ch * 8] = *(const v8h*)(A + (long)(m0 + row) * D_ + k0 + ch * 8);
      *(v8h*)&lds_bt[row * 48 + ch * 8] = *(const v8h*)(W + (long)(n0 + row) * D_ + k0 + ch * 8);
      *(v8h*)&lds_bt[(row + 64) * 48 + ch * 8] =
          *(const v8h*)(W + (long)(n0 + row + 64) * D_ + k0 + ch * 8);
    }
    __syncthreads();
    mma_step(lds_a, lds_bt, wy, wx, lane, c);
  }
#endif
}

#if HAVE_TDM
#define GEMM_LDS_BUFS 2
#else
#define GEMM_LDS_BUFS 1
#endif

// QKV projection: blockIdx.z selects (W_Q,Q)/(W_K,K)/(W_V,V); f16 out in [B,H,S,DK].
__global__ __launch_bounds__(256) void gemm_qkv(
    const _Float16* __restrict__ xh,
    const _Float16* __restrict__ wq, const _Float16* __restrict__ wk,
    const _Float16* __restrict__ wv,
    _Float16* __restrict__ Qh, _Float16* __restrict__ Kh, _Float16* __restrict__ Vh) {
  __shared__ _Float16 lds_a[GEMM_LDS_BUFS * 64 * 48];
  __shared__ _Float16 lds_bt[GEMM_LDS_BUFS * 128 * 48];
  const int z = blockIdx.z;
  const _Float16* W = (z == 0) ? wq : (z == 1) ? wk : wv;
  _Float16* dst = (z == 0) ? Qh : (z == 1) ? Kh : Vh;
  const int m0 = blockIdx.y * 64, n0 = blockIdx.x * 128;
  v8f c[2][2];
  gemm_core(xh, W, m0, n0, lds_a, lds_bt, c);

  const int lane = threadIdx.x & 31, wave = threadIdx.x >> 5;
  const int wy = wave >> 2, wx = wave & 3, hl = lane >> 4, nn = lane & 15;
#pragma unroll
  for (int ti = 0; ti < 2; ++ti)
#pragma unroll
    for (int tj = 0; tj < 2; ++tj)
#pragma unroll
      for (int r = 0; r < 8; ++r) {
        const int m = m0 + wy * 32 + ti * 16 + r + 8 * hl;   // token row
        const int n = n0 + wx * 32 + tj * 16 + nn;           // feature col
        const int bb = m >> 11, ss = m & (S_ - 1);
        const int hh = n >> 6, dk = n & 63;
        dst[((long)(bb * H_ + hh) * S_ + ss) * DK_ + dk] = (_Float16)c[ti][tj][r];
      }
}

// Output projection: fp32 straight to d_out.
__global__ __launch_bounds__(256) void gemm_oproj(
    const _Float16* __restrict__ Ah, const _Float16* __restrict__ wo,
    float* __restrict__ out) {
  __shared__ _Float16 lds_a[GEMM_LDS_BUFS * 64 * 48];
  __shared__ _Float16 lds_bt[GEMM_LDS_BUFS * 128 * 48];
  const int m0 = blockIdx.y * 64, n0 = blockIdx.x * 128;
  v8f c[2][2];
  gemm_core(Ah, wo, m0, n0, lds_a, lds_bt, c);

  const int lane = threadIdx.x & 31, wave = threadIdx.x >> 5;
  const int wy = wave >> 2, wx = wave & 3, hl = lane >> 4, nn = lane & 15;
#pragma unroll
  for (int ti = 0; ti < 2; ++ti)
#pragma unroll
    for (int tj = 0; tj < 2; ++tj)
#pragma unroll
      for (int r = 0; r < 8; ++r) {
        const int m = m0 + wy * 32 + ti * 16 + r + 8 * hl;
        const int n = n0 + wx * 32 + tj * 16 + nn;
        out[(long)m * D_ + n] = c[ti][tj][r];
      }
}

// ------------------------------------------------------------ flash attention
// Block = one (b,h) x 128 query rows; wave w owns rows [q0+16w, q0+16w+15].
// Per 32-key block: TDM stages K tile (padded stride 80), threads stage V
// transposed; scores via 4 WMMA, causal mask + online softmax in C-layout,
// P relaid through LDS, P*V via 4 WMMA.
__global__ __launch_bounds__(256) void flash_attn(
    const _Float16* __restrict__ Qh, const _Float16* __restrict__ Kh,
    const _Float16* __restrict__ Vh, _Float16* __restrict__ Ah) {
  __shared__ _Float16 lds_k[32 * 80];        // [key][dk], stride 80 = 128B + 32B pad
  __shared__ _Float16 lds_vt[64 * 48];       // [dk][key] transposed for B operand
  __shared__ _Float16 lds_p[8 * 16 * 48];    // per-wave 16x32 P tile (C->A relayout)

  const int tid = threadIdx.x, lane = tid & 31, wave = tid >> 5;
  const int hl = lane >> 4, nn = lane & 15;
  const int bh = blockIdx.y;                 // b*H + h
  const int b = bh >> 4, h = bh & 15;
  const long base = (long)bh * S_ * DK_;
  const int q0 = blockIdx.x * 128;
  const int qrow_w = q0 + wave * 16;
#if HAVE_TDM
  const unsigned lds_k_off = (unsigned)(size_t)&lds_k[0];
#endif

  // Q fragments for this wave's 16 rows (dk 0..31, 32..63), direct from global.
  const _Float16* qp = Qh + base + (long)qrow_w * DK_;
  const v16h aq0 = frag_a(qp, DK_, lane);
  const v16h aq1 = frag_a(qp + 32, DK_, lane);

  v8f o[4];
  {
    v8f z = {};
    o[0] = z; o[1] = z; o[2] = z; o[3] = z;
  }
  float mstate[8], lstate[8];
#pragma unroll
  for (int r = 0; r < 8; ++r) { mstate[r] = -1e38f; lstate[r] = 0.0f; }

  const int nj = (q0 + 128) >> 5;            // causal: key blocks up to the diagonal
  for (int j = 0; j < nj; ++j) {
    const int k0 = j * 32;
    __syncthreads();
    {
#if HAVE_TDM
      if (wave == 0)  // K tile: 32 keys x 64 dk, 32-DW rows + 8-DW pad -> stride 80
        tdm_load_2d(Kh + base + (long)k0 * DK_, lds_k_off, DK_, 32, DK_, 4, 7);
#endif
      // V tile transposed by all 256 threads (TDM cannot transpose)
      const int key = tid >> 3, ch = tid & 7;
#if !HAVE_TDM
      *(v8h*)&lds_k[key * 80 + ch * 8] =
          *(const v8h*)(Kh + base + (long)(k0 + key) * DK_ + ch * 8);
#endif
      const _Float16* vp = Vh + base + (long)(k0 + key) * DK_ + ch * 8;
      v8h vv = *(const v8h*)vp;
      if (j + 1 < nj) __builtin_prefetch(vp + 32 * DK_, 0, 0);  // next key block
#pragma unroll
      for (int i = 0; i < 8; ++i) lds_vt[(ch * 8 + i) * 48 + key] = vv[i];
#if HAVE_TDM
      if (wave == 0) __builtin_amdgcn_s_wait_tensorcnt(0);
#endif
    }
    __syncthreads();

    // scores: S = Q * K^T (two 16x16 key tiles, DK=64 -> 2 WMMA steps each)
    v8f s0 = {}, s1 = {};
    s0 = wmma16(aq0, frag_b(lds_k, 80, lane), s0);
    s0 = wmma16(aq1, frag_b(lds_k + 32, 80, lane), s0);
    s1 = wmma16(aq0, frag_b(lds_k + 16 * 80, 80, lane), s1);
    s1 = wmma16(aq1, frag_b(lds_k + 16 * 80 + 32, 80, lane), s1);

    // scale + causal mask + online softmax (rows in (r, lane-half); cols in lanes)
#pragma unroll
    for (int r = 0; r < 8; ++r) {
      const int row = qrow_w + r + 8 * hl;
      float v0 = s0[r] * 0.125f;             // 1/sqrt(64)
      float v1 = s1[r] * 0.125f;
      if (k0 + nn > row)      v0 = -1e30f;
      if (k0 + 16 + nn > row) v1 = -1e30f;
      float rm = fmaxf(v0, v1);
      rm = fmaxf(rm, __shfl_xor(rm, 1, 16));
      rm = fmaxf(rm, __shfl_xor(rm, 2, 16));
      rm = fmaxf(rm, __shfl_xor(rm, 4, 16));
      rm = fmaxf(rm, __shfl_xor(rm, 8, 16));
      const float mnew = fmaxf(mstate[r], rm);
      const float sc = __expf(mstate[r] - mnew);
      const float p0 = __expf(v0 - mnew);
      const float p1 = __expf(v1 - mnew);
      float rs = p0 + p1;
      rs += __shfl_xor(rs, 1, 16);
      rs += __shfl_xor(rs, 2, 16);
      rs += __shfl_xor(rs, 4, 16);
      rs += __shfl_xor(rs, 8, 16);
      lstate[r] = lstate[r] * sc + rs;
      mstate[r] = mnew;
      o[0][r] *= sc; o[1][r] *= sc; o[2][r] *= sc; o[3][r] *= sc;
      const int prow = r + 8 * hl;           // C-layout -> A-layout relay via LDS
      lds_p[(wave * 16 + prow) * 48 + nn]      = (_Float16)p0;
      lds_p[(wave * 16 + prow) * 48 + 16 + nn] = (_Float16)p1;
    }

    // O += P * V  (A = P 16x32, B = V from transposed LDS tile)
    const v16h ap = frag_a(lds_p + wave * 16 * 48, 48, lane);
#pragma unroll
    for (int t = 0; t < 4; ++t)
      o[t] = wmma16(ap, frag_b(lds_vt + t * 16 * 48, 48, lane), o[t]);
  }

  // epilogue: normalize, scatter to [b, s, h*64+dk] f16 for the O-proj GEMM
#pragma unroll
  for (int r = 0; r < 8; ++r) {
    const float inv = 1.0f / lstate[r];
    const int qrow = qrow_w + r + 8 * hl;
    const long orow = ((long)b * S_ + qrow) * D_ + h * DK_;
#pragma unroll
    for (int t = 0; t < 4; ++t)
      Ah[orow + t * 16 + nn] = (_Float16)(o[t][r] * inv);
  }
}

// ---------------------------------------------------------------- launcher
extern "C" void kernel_launch(void* const* d_in, const int* in_sizes, int n_in,
                              void* d_out, int out_size, void* d_ws, size_t ws_size,
                              hipStream_t stream) {
  (void)in_sizes; (void)n_in; (void)out_size; (void)ws_size;
  const float* x  = (const float*)d_in[0];
  const float* wq = (const float*)d_in[1];
  const float* wk = (const float*)d_in[2];
  const float* wv = (const float*)d_in[3];
  const float* wo = (const float*)d_in[4];
  float* out = (float*)d_out;

  const long XN = (long)B_ * S_ * D_;        // 8388608
  const long WN = (long)D_ * D_;             // 1048576
  _Float16* xh  = (_Float16*)d_ws;
  _Float16* wqh = xh + XN;
  _Float16* wkh = wqh + WN;
  _Float16* wvh = wkh + WN;
  _Float16* woh = wvh + WN;
  _Float16* Qh  = woh + WN;
  _Float16* Kh  = Qh + XN;
  _Float16* Vh  = Kh + XN;
  _Float16* Ah  = Vh + XN;                   // total ~92 MB of f16 workspace

  cvt_f32_f16<<<dim3(1024), 256, 0, stream>>>(x,  xh,  (int)(XN / 8));
  cvt_f32_f16<<<dim3(256),  256, 0, stream>>>(wq, wqh, (int)(WN / 8));
  cvt_f32_f16<<<dim3(256),  256, 0, stream>>>(wk, wkh, (int)(WN / 8));
  cvt_f32_f16<<<dim3(256),  256, 0, stream>>>(wv, wvh, (int)(WN / 8));
  cvt_f32_f16<<<dim3(256),  256, 0, stream>>>(wo, woh, (int)(WN / 8));

  // QKV: M=8192 tiles of 64 (y), N=1024 tiles of 128 (x), z = Q/K/V
  gemm_qkv<<<dim3(D_ / 128, (B_ * S_) / 64, 3), 256, 0, stream>>>(
      xh, wqh, wkh, wvh, Qh, Kh, Vh);

  // Flash attention: x = 16 query blocks of 128, y = B*H = 64
  flash_attn<<<dim3(S_ / 128, B_ * H_), 256, 0, stream>>>(Qh, Kh, Vh, Ah);

  // Output projection -> fp32 d_out
  gemm_oproj<<<dim3(D_ / 128, (B_ * S_) / 64), 256, 0, stream>>>(Ah, woh, out);
}
